// VQVAE_68375879352382
// MI455X (gfx1250) — matrix operand order repaired and checked by reading, not compiled, not run on hardware
//
#include <hip/hip_runtime.h>
#include <cstdint>

typedef __attribute__((ext_vector_type(16))) _Float16 v16h;
typedef __attribute__((ext_vector_type(8)))  float    v8f;

__device__ __forceinline__ v8f wmma_f16(v16h a, v16h b, v8f c) {
  // D = A(16x32 f16) * B(32x16 f16) + C(16x16 f32)
  return __builtin_amdgcn_wmma_f32_16x16x32_f16(
      /*neg_a=*/false, a, /*neg_b=*/false, b,
      /*c_mod=*/(short)0, c, /*reuse_a=*/false, /*reuse_b=*/false);
}

// Map k (0..31) to its LDS position: identity, so lane-half h reads
// k = 16h..16h+15 contiguously at offset h*16 (matches linear B fragments).
__device__ __forceinline__ int swz32(int k) {
  return k;
}

// f32 LDS fragment (16 contiguous floats) -> f16 vector
__device__ __forceinline__ v16h cvt16(const float* p) {
  v16h r;
#pragma unroll
  for (int e = 0; e < 16; ++e) r[e] = (_Float16)p[e];
  return r;
}

// ---- CDNA5 async global->LDS DMA (ASYNCcnt-tracked) -----------------------
// Generic shared-pointer low 32 bits == LDS byte offset.
__device__ __forceinline__ void async_b32(void* lds, const void* g) {
  unsigned lo = (unsigned)(uintptr_t)lds;
  unsigned long long ga = (unsigned long long)(uintptr_t)g;
  asm volatile("global_load_async_to_lds_b32 %0, %1, off"
               :: "v"(lo), "v"(ga) : "memory");
}
__device__ __forceinline__ void async_b128(void* lds, const void* g) {
  unsigned lo = (unsigned)(uintptr_t)lds;
  unsigned long long ga = (unsigned long long)(uintptr_t)g;
  asm volatile("global_load_async_to_lds_b128 %0, %1, off"
               :: "v"(lo), "v"(ga) : "memory");
}
__device__ __forceinline__ void wait_async_34() {
  asm volatile("s_wait_asynccnt 0x22" ::: "memory");
}
__device__ __forceinline__ void wait_async_1() {
  asm volatile("s_wait_asynccnt 0x1" ::: "memory");
}
__device__ __forceinline__ void wait_async_0() {
  asm volatile("s_wait_asynccnt 0x0" ::: "memory");
}

// ---------------------------------------------------------------------------
// Implicit-GEMM convolution, fully specialized per layer (all divides fold
// to shifts).  M = Cout, N = 64*Hgrid*Wgrid pixels, K = Ci*KH*KW.
// Weights are pre-converted/packed f16, flat idx = wbase + co*KTOT + kk.
// ---------------------------------------------------------------------------
template <int Ci, int Hin, int Win, int Co, int Hout, int Wout,
          int Hgrid, int Wgrid, int OS, int KH, int KW, int STR,
          int RELU, int KTOT>
__global__ void __launch_bounds__(256)
conv_wmma_kernel(const float* __restrict__ X, const _Float16* __restrict__ Wh,
                 const float* __restrict__ Bias, const float* __restrict__ zeroPad,
                 float* __restrict__ Y,
                 int pad_h, int pad_w, int oyb, int oxb, int wbase)
{
  constexpr int BM = 64, BN = 128, KC = 64;
  constexpr int HWg  = Hgrid * Wgrid;
  constexpr int Ntot = 64 * HWg;
  constexpr int KHW  = KH * KW;
  constexpr int nChunks = (KTOT + KC - 1) / KC;

  __shared__ alignas(32) _Float16 sW[2][BM * KC];  // f16 [m][(k/32)*32+swz32(k%32)]
  __shared__ alignas(16) float    sX[2][BN * KC];  // f32 [n][k] (B-frag linear)

  const int tid  = threadIdx.x;
  const int lane = tid & 31;
  const int wave = tid >> 5;
  const int wm   = wave >> 2;      // 0..1 : M sub-tile of 32
  const int wn   = wave & 3;       // 0..3 : N sub-tile of 32
  const int half = lane >> 4;
  const int l16  = lane & 15;
  const int tileN = blockIdx.x * BN;
  const int tileM = blockIdx.y * BM;
  const float*    zp  = zeroPad + (tid & 15);
  const _Float16* zph = (const _Float16*)zeroPad + ((tid & 7) << 3); // 16B-aligned

  v8f acc[2][2] = {};

  // one chunk of async global->LDS DMA: 2 b128 (weights) + 32 b32 (im2col)
  auto stage = [&](int buf, int k0) {
#pragma unroll
    for (int i = 0; i < 2; ++i) {
      int idx = tid + i * 256;                 // 512 x b128 = 4096 f16
      int m = idx >> 3, kg = (idx & 7) << 3;   // group of 8 k
      int co = tileM + m, kk = k0 + kg;
      const _Float16* src = zph;
      if (co < Co && kk < KTOT)                // KTOT % 8 == 0
        src = Wh + wbase + co * KTOT + kk;
      async_b128(&sW[buf][m * KC + ((kg >> 5) << 5) + swz32(kg & 31)], src);
    }
#pragma unroll
    for (int i = 0; i < 32; ++i) {
      int idx = tid + i * 256;                 // 8192 f32
      int n = idx >> 6, k = idx & 63;
      int kk = k0 + k;
      int ng = tileN + n;
      const float* src = zp;
      if (kk < KTOT && ng < Ntot) {
        int b   = ng / HWg;                    // constexpr divisor -> shift
        int rem = ng - b * HWg;
        int y = rem / Wgrid, x = rem - y * Wgrid;
        int ci = kk / KHW;
        int r  = kk - ci * KHW;
        int u  = r / KW, vv = r - u * KW;
        int ih = y * STR - pad_h + u;
        int iw = x * STR - pad_w + vv;
        if (ih >= 0 && ih < Hin && iw >= 0 && iw < Win)
          src = X + ((b * Ci + ci) * Hin + ih) * Win + iw;
      }
      async_b32(&sX[buf][n * KC + k], src);
    }
  };

  stage(0, 0);
  for (int ch = 0; ch < nChunks; ++ch) {
    const int cur = ch & 1;
    if (ch + 1 < nChunks) {
      stage(cur ^ 1, (ch + 1) * KC);   // DMA next chunk while computing
      wait_async_34();                  // in-order => chunk `cur` complete
    } else {
      wait_async_0();
    }
    __syncthreads();

    const _Float16* sWc = sW[cur];
    const float*    sXc = sX[cur];
#pragma unroll
    for (int kc = 0; kc < 2; ++kc) {
      v16h a0 = *(const v16h*)(sWc + (wm * 32 +      l16) * KC + kc * 32 + half * 16);
      v16h a1 = *(const v16h*)(sWc + (wm * 32 + 16 + l16) * KC + kc * 32 + half * 16);
      v16h b0 = cvt16(sXc + (wn * 32 +      l16) * KC + kc * 32 + half * 16);
      v16h b1 = cvt16(sXc + (wn * 32 + 16 + l16) * KC + kc * 32 + half * 16);
      acc[0][0] = wmma_f16(a0, b0, acc[0][0]);
      acc[0][1] = wmma_f16(a0, b1, acc[0][1]);
      acc[1][0] = wmma_f16(a1, b0, acc[1][0]);
      acc[1][1] = wmma_f16(a1, b1, acc[1][1]);
    }
    __syncthreads();   // LDS reads done before next stage overwrites `cur`
  }

  // ---- epilogue: bias + relu, scatter to NCHW ----
#pragma unroll
  for (int i = 0; i < 2; ++i) {
#pragma unroll
    for (int j = 0; j < 2; ++j) {
      int ng = tileN + wn * 32 + j * 16 + l16;
      if (ng < Ntot) {
        int b   = ng / HWg;
        int rem = ng - b * HWg;
        int y = rem / Wgrid, x = rem - y * Wgrid;
        int oh = y * OS + oyb, ow = x * OS + oxb;
#pragma unroll
        for (int r = 0; r < 8; ++r) {
          int co = tileM + wm * 32 + i * 16 + r + 8 * half;  // C-frag row map
          if (co < Co) {
            float v = acc[i][j][r] + Bias[co];
            if (RELU) v = fmaxf(v, 0.0f);
            Y[((b * Co + co) * Hout + oh) * Wout + ow] = v;
          }
        }
      }
    }
  }
}

// ---------------------------------------------------------------------------
// VQ argmin: scores = ||c||^2 - 2*z.c via WMMA.  codes = GEMM-M, pixels = N.
// f16 codebook blocks double-buffered via b128 DMA; raw v16h A-fragments.
// ---------------------------------------------------------------------------
__global__ void __launch_bounds__(256)
vq_argmin_kernel(const float* __restrict__ ZE, const float* __restrict__ CB,
                 const _Float16* __restrict__ CBh, int* __restrict__ IdxOut)
{
  constexpr int D = 64, HW = 1024, KCODES = 512;
  __shared__ alignas(32) _Float16 sC[2][32 * D];  // f16 code blocks [code][d]
  __shared__ alignas(16) float    sZ[128 * D];    // f32 [pixel][d]
  __shared__ float sCn[KCODES];

  const int tid  = threadIdx.x;
  const int lane = tid & 31;
  const int wave = tid >> 5;
  const int half = lane >> 4;
  const int l16  = lane & 15;
  const int pixelBase = blockIdx.x * 128;

  // async-stage z tile [pixel][d]
#pragma unroll
  for (int i = 0; i < 32; ++i) {
    int idx = tid + i * 256;        // 8192 = 128*64
    int d = idx >> 7, pl = idx & 127;
    int n = pixelBase + pl;
    int b = n >> 10, rem = n & 1023;
    async_b32(&sZ[pl * D + d], ZE + (b * D + d) * HW + rem);
  }
  // codebook norms in f32 (overlaps the DMA)
  for (int c = tid; c < KCODES; c += 256) {
    float s = 0.f;
#pragma unroll
    for (int d = 0; d < D; ++d) { float v = CB[c * D + d]; s += v * v; }
    sCn[c] = s;
  }
  wait_async_0();
  __syncthreads();

  // per-wave B fragments (fixed for whole code scan): 16 pixels, K = 64
  const int px = wave * 16 + l16;
  v16h bf0 = cvt16(sZ + px * D +  0 + half * 16);
  v16h bf1 = cvt16(sZ + px * D + 32 + half * 16);

  // stage a 32-code f16 block: 2048 f16 = 256 x b128 -> 1/thread
  auto stageC = [&](int buf, int cb) {
    int m = tid >> 3, kg = (tid & 7) << 3;
    async_b128(&sC[buf][m * D + ((kg >> 5) << 5) + swz32(kg & 31)],
               CBh + (cb * 32 + m) * D + kg);
  };

  float best = 3.4e38f;
  int  bidx = 0;
  stageC(0, 0);
  for (int cb = 0; cb < KCODES / 32; ++cb) {
    const int cur = cb & 1;
    if (cb + 1 < KCODES / 32) {
      stageC(cur ^ 1, cb + 1);
      wait_async_1();                  // current block complete
    } else {
      wait_async_0();
    }
    __syncthreads();

    const _Float16* sCc = sC[cur];
    v8f dot[2] = {};
#pragma unroll
    for (int i = 0; i < 2; ++i) {
      v16h a0 = *(const v16h*)(sCc + (i * 16 + l16) * D +  0 + half * 16);
      v16h a1 = *(const v16h*)(sCc + (i * 16 + l16) * D + 32 + half * 16);
      dot[i] = wmma_f16(a0, bf0, dot[i]);
      dot[i] = wmma_f16(a1, bf1, dot[i]);
    }
    // running argmin over the 16 codes this lane sees (ascending => first-min)
#pragma unroll
    for (int i = 0; i < 2; ++i) {
#pragma unroll
      for (int r = 0; r < 8; ++r) {
        int code = cb * 32 + i * 16 + r + 8 * half;
        float dist = sCn[code] - 2.0f * dot[i][r];
        if (dist < best) { best = dist; bidx = code; }
      }
    }
    __syncthreads();   // reads done before next stage overwrites
  }

  // wave halves hold complementary codes for the same pixel
  float ob = __shfl_xor(best, 16, 32);
  int   oi = __shfl_xor(bidx, 16, 32);
  if (ob < best || (ob == best && oi < bidx)) { best = ob; bidx = oi; }
  if (lane < 16) IdxOut[pixelBase + wave * 16 + l16] = bidx;
}

// ---------------------------------------------------------------------------
// Gather z_q (NCHW) + accumulate sum of (z_q - z)^2
// ---------------------------------------------------------------------------
__global__ void __launch_bounds__(256)
vq_gather_loss_kernel(const float* __restrict__ ZE, const float* __restrict__ CB,
                      const int* __restrict__ Idx, float* __restrict__ ZQ,
                      float* __restrict__ lossAcc)
{
  constexpr int D = 64, HW = 1024;
  __shared__ float red[256];
  int base = (blockIdx.x * 256 + threadIdx.x) * 4;
  float s = 0.f;
#pragma unroll
  for (int t = 0; t < 4; ++t) {
    int e = base + t;
    int n = e >> 6, d = e & 63;
    int b = n >> 10, rem = n & 1023;
    float z  = ZE[(b * D + d) * HW + rem];
    float zq = CB[Idx[n] * D + d];
    ZQ[(b * D + d) * HW + rem] = zq;
    float df = zq - z;
    s += df * df;
  }
  red[threadIdx.x] = s;
  __syncthreads();
  for (int off = 128; off > 0; off >>= 1) {
    if (threadIdx.x < off) red[threadIdx.x] += red[threadIdx.x + off];
    __syncthreads();
  }
  if (threadIdx.x == 0) atomicAdd(lossAcc, red[0]);
}

// ---------------------------------------------------------------------------
// One-time prep: f32 -> f16 weight/codebook conversion and tconv parity pack
// ---------------------------------------------------------------------------
__global__ void cvt_f16_kernel(const float* __restrict__ s,
                               _Float16* __restrict__ d, int n) {
  int i = blockIdx.x * 256 + threadIdx.x;
  if (i < n) d[i] = (_Float16)s[i];
}

// w[Co][Ci][4][4] -> d[par][Co][Ci*4] with kh=ohp+2u, kw=owp+2v, kk=ci*4+u*2+v
__global__ void pack_tconv_kernel(const float* __restrict__ w,
                                  _Float16* __restrict__ d, int Co, int Ci) {
  int K = Ci * 4;
  int total = 4 * Co * K;
  int i = blockIdx.x * 256 + threadIdx.x;
  if (i >= total) return;
  int par = i / (Co * K);
  int rem = i - par * (Co * K);
  int o = rem / K, kk = rem - o * K;
  int ci = kk >> 2, u = (kk >> 1) & 1, v = kk & 1;
  int ohp = par >> 1, owp = par & 1;
  d[i] = (_Float16)w[((o * Ci + ci) * 4 + (ohp + 2 * u)) * 4 + (owp + 2 * v)];
}

__global__ void init_kernel(float* lossAcc, float* zp) {
  if (threadIdx.x == 0) *lossAcc = 0.0f;
  zp[threadIdx.x] = 0.0f;     // 64-float zero pad for OOB async redirects
}

__global__ void vq_finalize_kernel(const float* __restrict__ lossAcc,
                                   float* __restrict__ outLoss)
{
  // vq_loss = codebook_loss + 0.25*commit_loss = 1.25 * MSE
  *outLoss = 1.25f * (*lossAcc) / 4194304.0f;   // N*D = 65536*64
}

// ---------------------------------------------------------------------------
// Host side
// ---------------------------------------------------------------------------
template <int Ci, int Hin, int Win, int Co, int Hout, int Wout,
          int Hgrid, int Wgrid, int OS, int KH, int KW, int STR,
          int RELU, int KTOT>
static void launch_conv(const float* X, const _Float16* Wh, const float* Bi,
                        const float* zp, float* Y, int pad_h, int pad_w,
                        int oyb, int oxb, int wbase, hipStream_t s)
{
  constexpr int Ntot = 64 * Hgrid * Wgrid;
  dim3 grid(Ntot / 128, (Co + 63) / 64);
  conv_wmma_kernel<Ci, Hin, Win, Co, Hout, Wout, Hgrid, Wgrid, OS, KH, KW, STR,
                   RELU, KTOT>
      <<<grid, 256, 0, s>>>(X, Wh, Bi, zp, Y, pad_h, pad_w, oyb, oxb, wbase);
}

extern "C" void kernel_launch(void* const* d_in, const int* in_sizes, int n_in,
                              void* d_out, int out_size, void* d_ws, size_t ws_size,
                              hipStream_t stream)
{
  const float* x      = (const float*)d_in[0];
  const float* enc_w1 = (const float*)d_in[1];
  const float* enc_b1 = (const float*)d_in[2];
  const float* enc_w2 = (const float*)d_in[3];
  const float* enc_b2 = (const float*)d_in[4];
  const float* enc_w3 = (const float*)d_in[5];
  const float* enc_b3 = (const float*)d_in[6];
  const float* cbook  = (const float*)d_in[7];
  const float* dec_w1 = (const float*)d_in[8];
  const float* dec_b1 = (const float*)d_in[9];
  const float* dec_w2 = (const float*)d_in[10];
  const float* dec_b2 = (const float*)d_in[11];
  const float* dec_w3 = (const float*)d_in[12];
  const float* dec_b3 = (const float*)d_in[13];

  float* outF = (float*)d_out;
  const size_t XR = (size_t)64 * 3 * 128 * 128;           // 3,145,728
  float* x_recon  = outF;
  float* vq_loss  = outF + XR;
  int*   indices  = (int*)(outF + XR + 1);                // 65,536 int32

  char* ws = (char*)d_ws;
  float* h1      = (float*)(ws);                           // 64*128*64*64 (reused as g2)
  float* h2      = (float*)(ws + 134217728ull);            // 64*128*32*32 (reused as g1)
  float* z_e     = (float*)(ws + 167772160ull);            // 64*64*32*32
  float* zq      = (float*)(ws + 184549376ull);            // 64*64*32*32
  float* lossAcc = (float*)(ws + 201326592ull);
  float* zeroPad = (float*)(ws + 201326848ull);            // 64 floats, 16B aligned
  _Float16* wf1  = (_Float16*)(ws + 201327104ull);         // 128*48
  _Float16* wf2  = (_Float16*)(ws + 201339392ull);         // 128*2048
  _Float16* wf3  = (_Float16*)(ws + 201863680ull);         // 64*128
  _Float16* wfd1 = (_Float16*)(ws + 201880064ull);         // 128*64
  _Float16* wfd2 = (_Float16*)(ws + 201896448ull);         // 4*128*512
  _Float16* wfd3 = (_Float16*)(ws + 202420736ull);         // 4*3*512
  _Float16* cbh  = (_Float16*)(ws + 202433024ull);         // 512*64

  init_kernel<<<1, 64, 0, stream>>>(lossAcc, zeroPad);

  // ---- one-time f16 weight conversion / tconv parity packing ----
  cvt_f16_kernel<<<(6144   + 255) / 256, 256, 0, stream>>>(enc_w1, wf1, 6144);
  cvt_f16_kernel<<<(262144 + 255) / 256, 256, 0, stream>>>(enc_w2, wf2, 262144);
  cvt_f16_kernel<<<(8192   + 255) / 256, 256, 0, stream>>>(enc_w3, wf3, 8192);
  cvt_f16_kernel<<<(8192   + 255) / 256, 256, 0, stream>>>(dec_w1, wfd1, 8192);
  cvt_f16_kernel<<<(32768  + 255) / 256, 256, 0, stream>>>(cbook, cbh, 32768);
  pack_tconv_kernel<<<(262144 + 255) / 256, 256, 0, stream>>>(dec_w2, wfd2, 128, 128);
  pack_tconv_kernel<<<(6144   + 255) / 256, 256, 0, stream>>>(dec_w3, wfd3, 3, 128);

  // ---- enc conv1: 3->128, k4 s2 p1, relu ----
  launch_conv<3, 128, 128, 128, 64, 64, 64, 64, 1, 4, 4, 2, 1, 48>(
      x, wf1, enc_b1, zeroPad, h1, 1, 1, 0, 0, 0, stream);

  // ---- enc conv2: 128->128, k4 s2 p1, relu ----
  launch_conv<128, 64, 64, 128, 32, 32, 32, 32, 1, 4, 4, 2, 1, 2048>(
      h1, wf2, enc_b2, zeroPad, h2, 1, 1, 0, 0, 0, stream);

  // ---- enc conv3: 1x1 128->64 ----
  launch_conv<128, 32, 32, 64, 32, 32, 32, 32, 1, 1, 1, 1, 0, 128>(
      h2, wf3, enc_b3, zeroPad, z_e, 0, 0, 0, 0, 0, stream);

  // ---- VQ: argmin + gather + loss ----
  vq_argmin_kernel<<<512, 256, 0, stream>>>(z_e, cbook, cbh, indices);
  vq_gather_loss_kernel<<<4096, 256, 0, stream>>>(z_e, cbook, indices, zq, lossAcc);
  vq_finalize_kernel<<<1, 1, 0, stream>>>(lossAcc, vq_loss);

  // ---- dec conv1: 1x1 64->128, relu (g1 reuses h2 buffer) ----
  launch_conv<64, 32, 32, 128, 32, 32, 32, 32, 1, 1, 1, 1, 1, 64>(
      zq, wfd1, dec_b1, zeroPad, h2, 0, 0, 0, 0, 0, stream);

  // ---- dec tconv1: 128->128 up2 -> 4 parity-class k2/s1 convs (g2 = h1) ----
  for (int ohp = 0; ohp < 2; ++ohp)
    for (int owp = 0; owp < 2; ++owp)
      launch_conv<128, 32, 32, 128, 64, 64, 32, 32, 2, 2, 2, 1, 1, 512>(
          h2, wfd2, dec_b2, zeroPad, h1,
          1 - ohp, 1 - owp, ohp, owp, (ohp * 2 + owp) * 128 * 512, stream);

  // ---- dec tconv2: 128->3 up2 -> 4 parity-class convs, writes x_recon ----
  for (int ohp = 0; ohp < 2; ++ohp)
    for (int owp = 0; owp < 2; ++owp)
      launch_conv<128, 64, 64, 3, 128, 128, 64, 64, 2, 2, 2, 1, 0, 512>(
          h1, wfd3, dec_b3, zeroPad, x_recon,
          1 - ohp, 1 - owp, ohp, owp, (ohp * 2 + owp) * 3 * 512, stream);
}
